// Decoder_Select_79328045957528
// MI455X (gfx1250) — matrix operand order repaired and checked by reading, not compiled, not run on hardware
//
#include <hip/hip_runtime.h>
#include <hip/hip_bf16.h>
#include <math.h>

// ---------------- problem constants ----------------
#define NN   16          // B*S
#define BN   128
#define CK   100
#define NC   52
#define SP   5200        // CK*NC
#define IC   256
#define NF   2450
#define NFP  2496        // padded frame count: 39 blocks * 64 columns
#define FB4  39          // f-blocks of 64 columns
#define KS   16
#define TT   19608       // KS + ST*(NF-1)
#define XW   72          // padded conv row width (x shifted by +2)
#define CW   64          // padded CAT width
#define OUT_WAVS (8ull*2*3*19608)   // 941184

typedef __attribute__((ext_vector_type(16))) _Float16 v16h;
typedef __attribute__((ext_vector_type(8)))  _Float16 h8;
typedef __attribute__((ext_vector_type(8)))  float    v8f;
typedef __attribute__((ext_vector_type(4)))  float    f4;

// ---------------- WMMA helpers (gfx1250 wave32) ----------------
__device__ __forceinline__ v8f wmma16(v16h a, v16h b, v8f c) {
  return __builtin_amdgcn_wmma_f32_16x16x32_f16(false, a, false, b, (short)0, c, false, false);
}

__device__ __forceinline__ v16h combine(h8 lo, h8 hi) {
  v16h r;
#pragma unroll
  for (int e = 0; e < 8; ++e) { r[e] = lo[e]; r[8 + e] = hi[e]; }
  return r;
}

// A operand: weights pre-packed in exact WMMA lane order -> 2 x b128 per lane
__device__ __forceinline__ v16h load_a_p(const _Float16* __restrict__ A,
                                         int mt, int kt, int ktiles, int lane) {
  const h8* q = (const h8*)(A + ((((size_t)mt * ktiles + kt) * 32 + lane) << 4));
  return combine(q[0], q[1]);
}

// B operand from channel-contiguous activations X[col][c] (ldc = channels), no predication
__device__ __forceinline__ v16h load_b_fc(const _Float16* __restrict__ X, int ldc,
                                          int k0, int f0, int lane) {
  const h8* q = (const h8*)(X + (size_t)(f0 + (lane & 15)) * ldc + k0 + ((lane >> 4) & 1) * 8);
  return combine(q[0], q[2]);
}

// ---------------- packing kernels ----------------
// Generic: pack f32 weight (element (m,k) at src[base + m*sm + k*sk]) into WMMA-A lane order.
__global__ void k_pack_a(const float* __restrict__ src, _Float16* __restrict__ dst,
                         int M, int K, int sm, int sk, int base) {
  int i = blockIdx.x * blockDim.x + threadIdx.x;
  if (i >= M * K) return;
  int e = i & 15;
  int lane = (i >> 4) & 31;
  int tile = i >> 9;
  int ktiles = K >> 5;
  int kt = tile % ktiles, mt = tile / ktiles;
  int m = mt * 16 + (lane & 15);
  int hi = lane >> 4;
  int v = e >> 1, kk = e & 1;
  int kb = (v < 4) ? (2 * v) : (16 + 2 * (v - 4));
  int k = kt * 32 + kb + hi * 8 + kk;
  dst[i] = (_Float16)src[base + m * sm + k * sk];
}

// xh[n][y][xp(72)][c] f16, x = xp-2 with explicit zero padding (conv zero-pad semantics)
__global__ void k_pack_x(const float* __restrict__ o0, const float* __restrict__ o1,
                         _Float16* __restrict__ xh) {
  long i = (long)blockIdx.x * blockDim.x + threadIdx.x;
  if (i >= (long)NN * CK * XW * BN) return;
  int c = (int)(i & (BN - 1));
  long r = i >> 7;
  int xp = (int)(r % XW); r /= XW;
  int y = (int)(r % CK);
  int n = (int)(r / CK);
  int x = xp - 2;
  float v = 0.f;
  if (x >= 0 && x < NC) {
    const float* src = (n & 1) ? o1 : o0;
    v = src[((long)(n >> 1) * BN + c) * SP + y * NC + x];
  }
  xh[i] = (_Float16)v;
}

// mixture_w [8][ic][f] -> mixwT [8][fp(2496)][ic] f32 (zero in pad columns)
__global__ void k_pack_mixT(const float* __restrict__ src, float* __restrict__ dst) {
  long i = (long)blockIdx.x * blockDim.x + threadIdx.x;
  if (i >= 8L * NFP * IC) return;
  int c = (int)(i & (IC - 1));
  long r = i >> 8;
  int f = (int)(r % NFP);
  int s = (int)(r / NFP);
  dst[i] = (f < NF) ? src[((size_t)s * IC + c) * NF + f] : 0.f;
}

// fold-before-GEMM: xfold[n][f][c] = prelu(x[..,f%50,f/50+2]) + prelu(x[..,f%50+50,f/50+1])
// valid (in-bounds gather) for all f < NFP: j <= 51 < 52, t <= 99 < 100
__global__ void k_fold_prelu(const float* __restrict__ o0, const float* __restrict__ o1,
                             const float* __restrict__ aptr, _Float16* __restrict__ xf) {
  long i = (long)blockIdx.x * blockDim.x + threadIdx.x;
  if (i >= (long)NN * NFP * BN) return;
  int c = (int)(i & (BN - 1));
  long r = i >> 7;
  int f = (int)(r % NFP);
  int n = (int)(r / NFP);
  float a = *aptr;
  int j1 = f / 50 + 2, t1 = f % 50;
  int j2 = f / 50 + 1, t2 = f % 50 + 50;
  const float* src = (n & 1) ? o1 : o0;
  const float* base = src + ((long)(n >> 1) * BN + c) * SP;
  float v1 = base[t1 * NC + j1]; v1 = v1 > 0.f ? v1 : a * v1;
  float v2 = base[t2 * NC + j2]; v2 = v2 > 0.f ? v2 : a * v2;
  xf[i] = (_Float16)(v1 + v2);
}

// ---------------- selector: implicit-GEMM 3x3 conv (two branches) ----------------
// unconditional shifted load from padded image
__device__ __forceinline__ v16h load_b_xp(const _Float16* __restrict__ xn,
                                          int c0, int yy, int xp0, int lane) {
  const h8* q = (const h8*)(xn + ((size_t)yy * XW + xp0 + (lane & 15)) * BN +
                            c0 + ((lane >> 4) & 1) * 8);
  return combine(q[0], q[2]);
}

// grid: (y*4+xt [400], oc_tile [4], n [16]); block 32; cat[n][y][64][oc(64)] f16
__global__ void k_conv_sel(const _Float16* __restrict__ W1p, const float* __restrict__ b1b,
                           const _Float16* __restrict__ W2p, const float* __restrict__ b2b,
                           const _Float16* __restrict__ XH, _Float16* __restrict__ CAT) {
  int lane = threadIdx.x;
  int bx = blockIdx.x; int y = bx >> 2; int x0 = (bx & 3) * 16;
  int octile = blockIdx.y; int n = blockIdx.z;
  int branch = octile >> 1; int dil = branch + 1;
  const _Float16* Wp = branch ? W2p : W1p;   // per-tap packed: [tap][2 mt][4 kt][32][16]
  const float* bias = branch ? b2b : b1b;
  int mt = octile & 1;
  const _Float16* xn = XH + (size_t)n * CK * XW * BN;
  v8f acc = {};
  for (int ky = 0; ky < 3; ++ky) {
    int yy = y + (ky - 1) * dil;
    if (yy < 0 || yy >= CK) continue;
    for (int kx = 0; kx < 3; ++kx) {
      int xp0 = x0 + (kx - 1) * dil + 2;      // [0, 52] -> in padded row
      const _Float16* Wt = Wp + (size_t)(ky * 3 + kx) * (32 * BN);
#pragma unroll
      for (int kt = 0; kt < 4; ++kt) {
        v16h a = load_a_p(Wt, mt, kt, 4, lane);
        v16h b = load_b_xp(xn, kt * 32, yy, xp0, lane);
        acc = wmma16(a, b, acc);
      }
    }
  }
  int col = lane & 15, hi = (lane >> 4) & 1;
  int ocl = mt * 16 + hi * 8;                 // within branch (0..31)
  h8 vs;
#pragma unroll
  for (int r = 0; r < 8; ++r) vs[r] = (_Float16)(acc[r] + bias[ocl + r]);
  *(h8*)(CAT + (((size_t)n * CK + y) * CW + x0 + col) * 64 + octile * 16 + hi * 8) = vs;
}

// grid (oc [64], n [16]) x 256 threads: per-channel mean(h), mean(relu(h)) after 1x1 fusion
__global__ void k_sel_reduce(const _Float16* __restrict__ CAT,
                             const float* __restrict__ fw, const float* __restrict__ fb,
                             float* __restrict__ meanH, float* __restrict__ meanR) {
  int oc = blockIdx.x, n = blockIdx.y, tid = threadIdx.x;
  __shared__ float s1[256], s2[256];
  const _Float16* cn = CAT + (size_t)n * CK * CW * 64;
  float a1 = 0.f, a2 = 0.f;
  for (int p = tid; p < SP; p += 256) {
    int y = p / NC, x = p % NC;
    const _Float16* cp = cn + ((size_t)y * CW + x) * 64;
    float hv = fb[oc];
#pragma unroll
    for (int c = 0; c < 64; ++c) hv += fw[oc * 64 + c] * (float)cp[c];
    a1 += hv; a2 += fmaxf(hv, 0.f);
  }
  s1[tid] = a1; s2[tid] = a2; __syncthreads();
  for (int w = 128; w > 0; w >>= 1) {
    if (tid < w) { s1[tid] += s1[tid + w]; s2[tid] += s2[tid + w]; }
    __syncthreads();
  }
  if (tid == 0) {
    meanH[n * 64 + oc] = s1[0] / (float)SP;
    meanR[n * 64 + oc] = s2[0] / (float)SP;
  }
}

// SE + classifier head; relu(h*s) == s*relu(h) since s=sigmoid(..)>0
__global__ void k_sel_head(const float* __restrict__ meanH, const float* __restrict__ meanR,
                           const float* __restrict__ se1, const float* __restrict__ se2,
                           const float* __restrict__ fcw, const float* __restrict__ fcb,
                           float* __restrict__ sel_out) {
  int n = threadIdx.x;
  if (n >= NN) return;
  float t1[16];
#pragma unroll
  for (int i = 0; i < 16; ++i) {
    float v = 0.f;
    for (int c = 0; c < 64; ++c) v += meanH[n * 64 + c] * se1[i * 64 + c];
    t1[i] = fmaxf(v, 0.f);
  }
  float pooled[64];
  for (int j = 0; j < 64; ++j) {
    float v = 0.f;
#pragma unroll
    for (int i = 0; i < 16; ++i) v += t1[i] * se2[j * 16 + i];
    float s = 1.f / (1.f + expf(-v));
    pooled[j] = s * meanR[n * 64 + j];
  }
  for (int k = 0; k < 2; ++k) {
    float v = fcb[k];
    for (int j = 0; j < 64; ++j) v += pooled[j] * fcw[k * 64 + j];
    sel_out[n * 2 + k] = v;
  }
}

// ---------------- decoder GEMMs (NT=4: one A tile x four f tiles, no predication) --------
// h2[n][f][o] = fo_w @ xfold + 2*fo_b ; grid (39, O/16, 16), block 32
__global__ void k_gemm_h2(const _Float16* __restrict__ Wp, const float* __restrict__ bias,
                          const _Float16* __restrict__ X, _Float16* __restrict__ H2, int O) {
  int lane = threadIdx.x;
  int f0 = blockIdx.x * 64, mt = blockIdx.y, n = blockIdx.z;
  const _Float16* Bm = X + (size_t)n * NFP * BN;
  v8f z8 = {};
  v8f acc[4] = {z8, z8, z8, z8};
#pragma unroll
  for (int kt = 0; kt < 4; ++kt) {
    v16h a = load_a_p(Wp, mt, kt, 4, lane);
#pragma unroll
    for (int t = 0; t < 4; ++t) {
      v16h b = load_b_fc(Bm, BN, kt * 32, f0 + t * 16, lane);
      acc[t] = wmma16(a, b, acc[t]);
    }
  }
  int col = lane & 15, hi = (lane >> 4) & 1;
  int o0 = mt * 16 + hi * 8;
  _Float16* out = H2 + (size_t)n * NFP * O + o0;
#pragma unroll
  for (int t = 0; t < 4; ++t) {
    int f = f0 + t * 16 + col;
    h8 vs;
#pragma unroll
    for (int r = 0; r < 8; ++r) vs[r] = (_Float16)(acc[t][r] + 2.f * bias[o0 + r]);
    *(h8*)(out + (size_t)f * O) = vs;
  }
}

// og[bb][f][c'] = tanh(no@h2+nb) * sigmoid(ng@h2+gb); grid (39, 8, 16*nsrc)
__global__ void k_gemm_gate(const _Float16* __restrict__ Wnp, const float* __restrict__ bno,
                            const _Float16* __restrict__ Wgp, const float* __restrict__ bng,
                            const _Float16* __restrict__ H2, _Float16* __restrict__ OG,
                            int nsrc) {
  int lane = threadIdx.x;
  int f0 = blockIdx.x * 64, mt = blockIdx.y, bb = blockIdx.z;
  int n = bb / nsrc, s = bb % nsrc;
  int O = nsrc * BN;
  const _Float16* Bm = H2 + (size_t)n * NFP * O + s * BN;
  v8f z8 = {};
  v8f ao[4] = {z8, z8, z8, z8};
  v8f ag[4] = {z8, z8, z8, z8};
#pragma unroll
  for (int kt = 0; kt < 4; ++kt) {
    v16h b[4];
#pragma unroll
    for (int t = 0; t < 4; ++t) b[t] = load_b_fc(Bm, O, kt * 32, f0 + t * 16, lane);
    v16h aO = load_a_p(Wnp, mt, kt, 4, lane);
#pragma unroll
    for (int t = 0; t < 4; ++t) ao[t] = wmma16(aO, b[t], ao[t]);
    v16h aG = load_a_p(Wgp, mt, kt, 4, lane);
#pragma unroll
    for (int t = 0; t < 4; ++t) ag[t] = wmma16(aG, b[t], ag[t]);
  }
  int col = lane & 15, hi = (lane >> 4) & 1;
  int c0 = mt * 16 + hi * 8;
  _Float16* out = OG + (size_t)bb * NFP * BN + c0;
#pragma unroll
  for (int t = 0; t < 4; ++t) {
    int f = f0 + t * 16 + col;
    h8 vs;
#pragma unroll
    for (int r = 0; r < 8; ++r) {
      float ov = tanhf(ao[t][r] + bno[c0 + r]);
      float gv = 1.f / (1.f + expf(-(ag[t][r] + bng[c0 + r])));
      vs[r] = (_Float16)(ov * gv);
    }
    *(h8*)(out + (size_t)f * BN) = vs;
  }
}

// src[bb][f][ic] = relu(mn@og) * mixwT[n>>1][f][ic]; grid (39, 16, 16*nsrc)
__global__ void k_gemm_src(const _Float16* __restrict__ Wp, const _Float16* __restrict__ OG,
                           const float* __restrict__ mixT, _Float16* __restrict__ SRC,
                           int nsrc) {
  int lane = threadIdx.x;
  int f0 = blockIdx.x * 64, mt = blockIdx.y, bb = blockIdx.z;
  const _Float16* Bm = OG + (size_t)bb * NFP * BN;
  v8f z8 = {};
  v8f acc[4] = {z8, z8, z8, z8};
#pragma unroll
  for (int kt = 0; kt < 4; ++kt) {
    v16h a = load_a_p(Wp, mt, kt, 4, lane);
#pragma unroll
    for (int t = 0; t < 4; ++t) {
      v16h b = load_b_fc(Bm, BN, kt * 32, f0 + t * 16, lane);
      acc[t] = wmma16(a, b, acc[t]);
    }
  }
  int col = lane & 15, hi = (lane >> 4) & 1;
  int ic0 = mt * 16 + hi * 8;
  int n = bb / nsrc;
  const float* mp = mixT + (size_t)(n >> 1) * NFP * IC + ic0;
  _Float16* out = SRC + (size_t)bb * NFP * IC + ic0;
#pragma unroll
  for (int t = 0; t < 4; ++t) {
    int f = f0 + t * 16 + col;
    const f4* m4 = (const f4*)(mp + (size_t)f * IC);
    f4 mlo = m4[0], mhi = m4[1];
    h8 vs;
#pragma unroll
    for (int r = 0; r < 8; ++r) {
      float mv = (r < 4) ? mlo[r] : mhi[r - 4];
      vs[r] = (_Float16)(fmaxf(acc[t][r], 0.f) * mv);
    }
    *(h8*)(out + (size_t)f * IC) = vs;
  }
}

// frames[bb][f][k] = sum_c tc_w[c,k]*src[bb][f][c]; A = tc^T packed [16x256]; grid (39,1,NB)
__global__ void k_gemm_frames(const _Float16* __restrict__ Wp, const _Float16* __restrict__ SRC,
                              float* __restrict__ FR) {
  int lane = threadIdx.x;
  int f0 = blockIdx.x * 64, bb = blockIdx.z;
  const _Float16* Bm = SRC + (size_t)bb * NFP * IC;
  v8f z8 = {};
  v8f acc[4] = {z8, z8, z8, z8};
#pragma unroll
  for (int kt = 0; kt < 8; ++kt) {
    v16h a = load_a_p(Wp, 0, kt, 8, lane);
#pragma unroll
    for (int t = 0; t < 4; ++t) {
      v16h b = load_b_fc(Bm, IC, kt * 32, f0 + t * 16, lane);
      acc[t] = wmma16(a, b, acc[t]);
    }
  }
  int col = lane & 15, hi = (lane >> 4) & 1;
  float* out = FR + (size_t)bb * NFP * KS + hi * 8;
#pragma unroll
  for (int t = 0; t < 4; ++t) {
    int f = f0 + t * 16 + col;
    f4 lo, hh;
#pragma unroll
    for (int r = 0; r < 4; ++r) { lo[r] = acc[t][r]; hh[r] = acc[t][r + 4]; }
    f4* o4 = (f4*)(out + (size_t)f * KS);
    o4[0] = lo; o4[1] = hh;
  }
}

// final overlap-add (stride 8, kernel 16 => <=2 contributions) + per-sample decoder dispatch
__global__ void k_final(const float* __restrict__ FR0, const float* __restrict__ FR1,
                        const int* __restrict__ gt, float* __restrict__ out) {
  long i = (long)blockIdx.x * blockDim.x + threadIdx.x;
  if (i >= (long)OUT_WAVS) return;
  int t = (int)(i % TT); long r = i / TT;
  int slot = (int)(r % 3); r /= 3;
  int stage = (int)(r % 2); int b = (int)(r / 2);
  int d = gt[b] - 2; d = d < 0 ? 0 : (d > 1 ? 1 : d);
  int nsrc = 2 + d;
  float v = 0.f;
  if (slot < nsrc) {
    int bfull = (b * 2 + stage) * nsrc + slot;
    const float* fr = (d ? FR1 : FR0) + (size_t)bfull * NFP * KS;
    int f1 = t >> 3, k1 = t & 7;
    if (f1 < NF) v += fr[(size_t)f1 * KS + k1];
    int f0 = f1 - 1;
    if (f0 >= 0) v += fr[(size_t)f0 * KS + k1 + 8];
  }
  out[i] = v;
}

// ---------------- workspace layout (bytes) ----------------
#define H2B 2ull
constexpr size_t OFF_WB1  = 0;                                  // conv b1: 9 taps * 32*128
constexpr size_t OFF_WB2  = OFF_WB1 + 36864ull * H2B;
constexpr size_t OFF_WFO0 = OFF_WB2 + 36864ull * H2B;
constexpr size_t OFF_WFO1 = OFF_WFO0 + 32768ull * H2B;
constexpr size_t OFF_WNO0 = OFF_WFO1 + 49152ull * H2B;
constexpr size_t OFF_WNO1 = OFF_WNO0 + 16384ull * H2B;
constexpr size_t OFF_WNG0 = OFF_WNO1 + 16384ull * H2B;
constexpr size_t OFF_WNG1 = OFF_WNG0 + 16384ull * H2B;
constexpr size_t OFF_WMN0 = OFF_WNG1 + 16384ull * H2B;
constexpr size_t OFF_WMN1 = OFF_WMN0 + 32768ull * H2B;
constexpr size_t OFF_WTC0 = OFF_WMN1 + 32768ull * H2B;
constexpr size_t OFF_WTC1 = OFF_WTC0 + 4096ull * H2B;
constexpr size_t OFF_MEANH = OFF_WTC1 + 4096ull * H2B;
constexpr size_t OFF_MEANR = OFF_MEANH + 1024ull * 4;
constexpr size_t OFF_MWT   = OFF_MEANR + 1024ull * 4;           // 8*2496*256 f32
constexpr size_t OFF_FR0   = OFF_MWT + 8ull * NFP * IC * 4;
constexpr size_t OFF_FR1   = OFF_FR0 + 32ull * NFP * KS * 4;
constexpr size_t OFF_ARENA = OFF_FR1 + 48ull * NFP * KS * 4;
// decoder arena (both decoders reuse these; stream-sequential)
constexpr size_t A_XFOLD = OFF_ARENA;                           // 16*2496*128 f16
constexpr size_t A_H2    = A_XFOLD + 16ull * NFP * BN * H2B;    // up to 16*2496*384 f16
constexpr size_t A_OG    = A_H2 + 16ull * NFP * 384 * H2B;      // up to 48*2496*128 f16
constexpr size_t A_SRC   = A_OG + 48ull * NFP * BN * H2B;       // up to 48*2496*256 f16
// selector scratch aliases the decoder arena (selector fully completes first)
constexpr size_t A_XH  = OFF_ARENA;                             // 16*100*72*128 f16
constexpr size_t A_CAT = A_XH + (size_t)NN * CK * XW * BN * H2B;// 16*100*64*64 f16
// total ws needed ~167 MB

extern "C" void kernel_launch(void* const* d_in, const int* in_sizes, int n_in,
                              void* d_out, int out_size, void* d_ws, size_t ws_size,
                              hipStream_t stream) {
  const float* output0 = (const float*)d_in[0];
  const float* output1 = (const float*)d_in[1];
  const float* mixw    = (const float*)d_in[2];
  const int*   gt      = (const int*)d_in[3];
  const float* b1_w = (const float*)d_in[4];  const float* b1_b = (const float*)d_in[5];
  const float* b2_w = (const float*)d_in[6];  const float* b2_b = (const float*)d_in[7];
  const float* fus_w = (const float*)d_in[8]; const float* fus_b = (const float*)d_in[9];
  const float* se1 = (const float*)d_in[10];  const float* se2 = (const float*)d_in[11];
  const float* fc_w = (const float*)d_in[12]; const float* fc_b = (const float*)d_in[13];
  const float* prelu[2] = {(const float*)d_in[14], (const float*)d_in[23]};
  const float* fo_w[2] = {(const float*)d_in[15], (const float*)d_in[24]};
  const float* fo_b[2] = {(const float*)d_in[16], (const float*)d_in[25]};
  const float* no_w[2] = {(const float*)d_in[17], (const float*)d_in[26]};
  const float* no_b[2] = {(const float*)d_in[18], (const float*)d_in[27]};
  const float* ng_w[2] = {(const float*)d_in[19], (const float*)d_in[28]};
  const float* ng_b[2] = {(const float*)d_in[20], (const float*)d_in[29]};
  const float* mn_w[2] = {(const float*)d_in[21], (const float*)d_in[30]};
  const float* tc_w[2] = {(const float*)d_in[22], (const float*)d_in[31]};

  char* ws = (char*)d_ws;
  float* out = (float*)d_out;

  auto packA = [&](const float* s, size_t off, int M, int K, int sm, int sk, int base) {
    int n = M * K;
    k_pack_a<<<(n + 255) / 256, 256, 0, stream>>>(s, (_Float16*)(ws + off), M, K, sm, sk, base);
  };

  // ---- weight packing into WMMA-A lane order ----
  for (int tap = 0; tap < 9; ++tap) {   // conv weights: per-tap [32 x 128] GEMM operands
    packA(b1_w, OFF_WB1 + (size_t)tap * 32 * BN * H2B, 32, BN, 1152, 9, tap);
    packA(b2_w, OFF_WB2 + (size_t)tap * 32 * BN * H2B, 32, BN, 1152, 9, tap);
  }
  packA(fo_w[0], OFF_WFO0, 256, BN, BN, 1, 0);
  packA(fo_w[1], OFF_WFO1, 384, BN, BN, 1, 0);
  packA(no_w[0], OFF_WNO0, BN, BN, BN, 1, 0);
  packA(no_w[1], OFF_WNO1, BN, BN, BN, 1, 0);
  packA(ng_w[0], OFF_WNG0, BN, BN, BN, 1, 0);
  packA(ng_w[1], OFF_WNG1, BN, BN, BN, 1, 0);
  packA(mn_w[0], OFF_WMN0, IC, BN, BN, 1, 0);
  packA(mn_w[1], OFF_WMN1, IC, BN, BN, 1, 0);
  packA(tc_w[0], OFF_WTC0, 16, IC, 1, 16, 0);   // transposed: A[k_out, c] = tc_w[c*16 + k_out]
  packA(tc_w[1], OFF_WTC1, 16, IC, 1, 16, 0);

  {
    long nm = 8L * NFP * IC;
    k_pack_mixT<<<(unsigned)((nm + 255) / 256), 256, 0, stream>>>(mixw, (float*)(ws + OFF_MWT));
  }

  // ---- selector ----
  {
    long n = (long)NN * CK * XW * BN;
    k_pack_x<<<(unsigned)((n + 255) / 256), 256, 0, stream>>>(output0, output1,
                                                              (_Float16*)(ws + A_XH));
    k_conv_sel<<<dim3(400, 4, NN), 32, 0, stream>>>(
        (const _Float16*)(ws + OFF_WB1), b1_b,
        (const _Float16*)(ws + OFF_WB2), b2_b,
        (const _Float16*)(ws + A_XH), (_Float16*)(ws + A_CAT));
    k_sel_reduce<<<dim3(64, NN), 256, 0, stream>>>(
        (const _Float16*)(ws + A_CAT), fus_w, fus_b,
        (float*)(ws + OFF_MEANH), (float*)(ws + OFF_MEANR));
    k_sel_head<<<1, 32, 0, stream>>>(
        (const float*)(ws + OFF_MEANH), (const float*)(ws + OFF_MEANR),
        se1, se2, fc_w, fc_b, out + OUT_WAVS);
  }

  // ---- decoders (share arena; frames kept separate for final dispatch) ----
  const size_t offFO[2] = {OFF_WFO0, OFF_WFO1};
  const size_t offNO[2] = {OFF_WNO0, OFF_WNO1};
  const size_t offNG[2] = {OFF_WNG0, OFF_WNG1};
  const size_t offMN[2] = {OFF_WMN0, OFF_WMN1};
  const size_t offTC[2] = {OFF_WTC0, OFF_WTC1};
  const size_t offFR[2] = {OFF_FR0, OFF_FR1};

  for (int d = 0; d < 2; ++d) {
    int nsrc = 2 + d;
    int O = nsrc * BN;
    int NB = NN * nsrc;
    long nf = (long)NN * NFP * BN;
    k_fold_prelu<<<(unsigned)((nf + 255) / 256), 256, 0, stream>>>(
        output0, output1, prelu[d], (_Float16*)(ws + A_XFOLD));
    k_gemm_h2<<<dim3(FB4, O / 16, NN), 32, 0, stream>>>(
        (const _Float16*)(ws + offFO[d]), fo_b[d],
        (const _Float16*)(ws + A_XFOLD), (_Float16*)(ws + A_H2), O);
    k_gemm_gate<<<dim3(FB4, 8, NB), 32, 0, stream>>>(
        (const _Float16*)(ws + offNO[d]), no_b[d],
        (const _Float16*)(ws + offNG[d]), ng_b[d],
        (const _Float16*)(ws + A_H2), (_Float16*)(ws + A_OG), nsrc);
    k_gemm_src<<<dim3(FB4, 16, NB), 32, 0, stream>>>(
        (const _Float16*)(ws + offMN[d]), (const _Float16*)(ws + A_OG),
        (const float*)(ws + OFF_MWT), (_Float16*)(ws + A_SRC), nsrc);
    k_gemm_frames<<<dim3(FB4, 1, NB), 32, 0, stream>>>(
        (const _Float16*)(ws + offTC[d]), (const _Float16*)(ws + A_SRC),
        (float*)(ws + offFR[d]));
  }

  // ---- final overlap-add + decoder selection ----
  k_final<<<(unsigned)((OUT_WAVS + 255) / 256), 256, 0, stream>>>(
      (const float*)(ws + OFF_FR0), (const float*)(ws + OFF_FR1), gt, out);
}
